// GCNModel_48172353192007
// MI455X (gfx1250) — matrix operand order repaired and checked by reading, not compiled, not run on hardware
//
#include <hip/hip_runtime.h>
#include <hip/hip_bf16.h>

// ---------------------------------------------------------------------------
// GCN forward (2x GCNConv + linear) for MI455X (gfx1250, wave32, WMMA).
// GEMMs: v_wmma_f32_16x16x32_bf16, 16x64 register blocking per wave,
//        B-panel staged in LDS (ds_load path), A streamed from global/L2.
// Edge aggregation: L2-resident gather + global f32 atomics.
// ---------------------------------------------------------------------------

typedef __bf16 bf16_t;
typedef __attribute__((ext_vector_type(8)))  __bf16 v8bf;
typedef __attribute__((ext_vector_type(16))) __bf16 v16bf;
typedef __attribute__((ext_vector_type(8)))  float  v8f;

#define GCN_N   50000
#define GCN_E   800000
#define GCN_IN  256
#define GCN_H   512
#define GCN_OUT 64

#define BM_TILES 8      // 8 waves per block, one 16-row m-tile each
#define BN       64     // 64 output columns per block (4 n-tiles per wave)

static __device__ __forceinline__ bf16_t f2bf(float f) { return (bf16_t)f; }

// ------------------------- degree / norm ------------------------------------
__global__ void deg_init_kernel(float* __restrict__ deg, int n) {
    int i = blockIdx.x * blockDim.x + threadIdx.x;
    if (i < n) deg[i] = 1.0f;   // self-loop contributes 1 to every node
}

__global__ void deg_accum_kernel(const int* __restrict__ dst, float* __restrict__ deg, int e) {
    int stride = gridDim.x * blockDim.x;
    for (int i = blockIdx.x * blockDim.x + threadIdx.x; i < e; i += stride)
        atomicAdd(&deg[dst[i]], 1.0f);
}

__global__ void deg_rsqrt_kernel(float* __restrict__ deg, int n) {
    int i = blockIdx.x * blockDim.x + threadIdx.x;
    if (i < n) deg[i] = rsqrtf(deg[i]);   // deg >= 1 always
}

// ------------------------- bf16 packing -------------------------------------
__global__ void cvt_bf16_kernel(const float* __restrict__ x, bf16_t* __restrict__ y, int n) {
    int stride = gridDim.x * blockDim.x;
    for (int i = blockIdx.x * blockDim.x + threadIdx.x; i < n; i += stride)
        y[i] = f2bf(x[i]);
}

// W is [K, Nc] row-major; Wt is [Nc, K] row-major (bf16) so a WMMA B-fragment
// (16 contiguous K values of one output column per lane) is one 32B read.
__global__ void pack_wT_kernel(const float* __restrict__ W, bf16_t* __restrict__ Wt,
                               int K, int Nc) {
    int stride = gridDim.x * blockDim.x;
    int total = K * Nc;
    for (int i = blockIdx.x * blockDim.x + threadIdx.x; i < total; i += stride) {
        int k = i / Nc;
        int n = i % Nc;
        Wt[(size_t)n * K + k] = f2bf(W[i]);
    }
}

// ------------------------- WMMA GEMM ----------------------------------------
// C[M,Nc] = A[M,K] (bf16 row-major) * Bt[Nc,K]^T (bf16) [+ bias]
// Block: 256 threads (8 waves). Block tile: 128 rows x 64 cols.
// Wave tile: 16 rows x 64 cols (4 accumulators, shared A fragment).
// B panel (64 cols x K) staged once into LDS; fragment layouts used here
// (contiguous k order, same for A and B):
//   A 16x32 bf16, lane L (m=L%16, half=L/16):
//       a[j]   = A[m][k0 + 16*half + j]     (contiguous 32B, j < 16)
//   B 32x16 bf16, lane L (n=L%16, half=L/16):
//       b[j]   = Bt[n][k0 + 16*half + j]     (contiguous 32B)
//   C f32 16x16, lane L (n=L%16): c[r] -> row r + 8*(L/16)
__global__ __launch_bounds__(256)
void wmma_gemm_blocked_kernel(const bf16_t* __restrict__ A,
                              const bf16_t* __restrict__ Bt,
                              float* __restrict__ C,
                              const float* __restrict__ bias,
                              int M, int Nc, int K) {
    __shared__ bf16_t Bs[BN * GCN_H];           // 64 KB max (K <= 512)

    const int tid  = threadIdx.x;
    const int lane = tid & 31;
    const int wave = tid >> 5;
    const int nblocks = Nc / BN;
    const int mblk = blockIdx.x / nblocks;
    const int nblk = blockIdx.x % nblocks;
    const int baseN = nblk * BN;

    // cooperative stage: Bt[baseN .. baseN+63][0..K) -> LDS
    {
        const uint4* s = reinterpret_cast<const uint4*>(Bt + (size_t)baseN * K);
        uint4* d = reinterpret_cast<uint4*>(Bs);
        const int chunks = BN * K / 8;          // 8 bf16 per 16B chunk
        for (int c = tid; c < chunks; c += 256) d[c] = s[c];
    }
    __syncthreads();

    const int mtile = mblk * BM_TILES + wave;
    if (mtile < (M >> 4)) {                     // wave-uniform: EXEC all-1s inside
        const int baseM = mtile << 4;
        const int half  = lane >> 4;
        const int lm    = lane & 15;

        const bf16_t* arow  = A + (size_t)(baseM + lm) * K;
        const bf16_t* bbase = Bs + (size_t)lm * K + 16 * half;

        v8f acc[4] = {{}, {}, {}, {}};
        for (int k0 = 0; k0 < K; k0 += 32) {
            v8bf alo = *reinterpret_cast<const v8bf*>(arow + k0 + 16 * half);
            v8bf ahi = *reinterpret_cast<const v8bf*>(arow + k0 + 8 + 16 * half);
            v16bf a;
#pragma unroll
            for (int j = 0; j < 8; ++j) { a[j] = alo[j]; a[8 + j] = ahi[j]; }

            v16bf b0 = *reinterpret_cast<const v16bf*>(bbase + k0);
            v16bf b1 = *reinterpret_cast<const v16bf*>(bbase + (size_t)16 * K + k0);
            v16bf b2 = *reinterpret_cast<const v16bf*>(bbase + (size_t)32 * K + k0);
            v16bf b3 = *reinterpret_cast<const v16bf*>(bbase + (size_t)48 * K + k0);

            acc[0] = __builtin_amdgcn_wmma_f32_16x16x32_bf16(false, a, false, b0, (short)0, acc[0], false, false);
            acc[1] = __builtin_amdgcn_wmma_f32_16x16x32_bf16(false, a, false, b1, (short)0, acc[1], false, false);
            acc[2] = __builtin_amdgcn_wmma_f32_16x16x32_bf16(false, a, false, b2, (short)0, acc[2], false, false);
            acc[3] = __builtin_amdgcn_wmma_f32_16x16x32_bf16(false, a, false, b3, (short)0, acc[3], false, false);
        }

#pragma unroll
        for (int t = 0; t < 4; ++t) {
            const int n  = baseN + 16 * t + lm;
            const float bv = bias ? bias[n] : 0.0f;
#pragma unroll
            for (int r = 0; r < 8; ++r) {
                const int row = baseM + r + 8 * half;
                C[(size_t)row * Nc + n] = acc[t][r] + bv;
            }
        }
    }
}

// ------------------------- aggregation --------------------------------------
// agg[i][:] = xw[i][:] * dinv[i]^2      (self-loop term, also initializes)
__global__ void agg_init_kernel(const float* __restrict__ xw,
                                const float* __restrict__ dinv,
                                float* __restrict__ agg, int nNodes, int F) {
    int fq = F >> 2;
    int total = nNodes * fq;
    int stride = gridDim.x * blockDim.x;
    for (int t = blockIdx.x * blockDim.x + threadIdx.x; t < total; t += stride) {
        int i = t / fq;
        float di = dinv[i];
        float s = di * di;
        float4 v = reinterpret_cast<const float4*>(xw)[t];
        v.x *= s; v.y *= s; v.z *= s; v.w *= s;
        reinterpret_cast<float4*>(agg)[t] = v;
    }
}

// agg[dst][:] += xw[src][:] * (dinv[src]*dinv[dst])  over E edges
__global__ void edge_scatter_kernel(const float* __restrict__ xw,
                                    const int* __restrict__ src,
                                    const int* __restrict__ dst,
                                    const float* __restrict__ dinv,
                                    float* __restrict__ agg, int e, int F) {
    const int fq = F >> 2;
    const size_t total = (size_t)e * fq;
    const size_t stride = (size_t)gridDim.x * blockDim.x;
    for (size_t t = (size_t)blockIdx.x * blockDim.x + threadIdx.x; t < total; t += stride) {
        int ei = (int)(t / fq);
        int c  = (int)(t % fq);
        int s = src[ei];
        int d = dst[ei];
        float nrm = dinv[s] * dinv[d];
        float4 v = reinterpret_cast<const float4*>(xw + (size_t)s * F)[c];
        float* out = agg + (size_t)d * F + 4 * c;
        atomicAdd(out + 0, v.x * nrm);
        atomicAdd(out + 1, v.y * nrm);
        atomicAdd(out + 2, v.z * nrm);
        atomicAdd(out + 3, v.w * nrm);
    }
}

// h_bf16 = bf16(relu(agg + bias))
__global__ void finalize_kernel(const float* __restrict__ agg,
                                const float* __restrict__ bias,
                                bf16_t* __restrict__ out, int nNodes, int F) {
    int total = nNodes * F;
    int stride = gridDim.x * blockDim.x;
    for (int t = blockIdx.x * blockDim.x + threadIdx.x; t < total; t += stride) {
        int f = t & (F - 1);                    // F is a power of two (512)
        float v = agg[t] + bias[f];
        out[t] = f2bf(fmaxf(v, 0.0f));
    }
}

// ---------------------------------------------------------------------------
static inline size_t align256(size_t x) { return (x + 255) & ~(size_t)255; }

extern "C" void kernel_launch(void* const* d_in, const int* in_sizes, int n_in,
                              void* d_out, int out_size, void* d_ws, size_t ws_size,
                              hipStream_t stream) {
    (void)in_sizes; (void)n_in; (void)out_size; (void)ws_size;
    const int N = GCN_N, E = GCN_E, IN = GCN_IN, H = GCN_H, OUT = GCN_OUT;

    const float* x    = (const float*)d_in[0];
    const int*   eidx = (const int*)d_in[1];     // [2, E] flattened
    const int*   src  = eidx;
    const int*   dst  = eidx + E;
    const float* W1 = (const float*)d_in[2];
    const float* b1 = (const float*)d_in[3];
    const float* W2 = (const float*)d_in[4];
    const float* b2 = (const float*)d_in[5];
    const float* Wl = (const float*)d_in[6];
    const float* bl = (const float*)d_in[7];
    float* out = (float*)d_out;

    // workspace carve-up
    char* ws = (char*)d_ws;
    size_t off = 0;
    float* dinv = (float*)(ws + off);       off += align256((size_t)N * 4);
    bf16_t* xbf = (bf16_t*)(ws + off);      off += align256((size_t)N * IN * 2);
    bf16_t* hbf = (bf16_t*)(ws + off);      off += align256((size_t)N * H * 2);
    float* gout = (float*)(ws + off);       off += align256((size_t)N * H * 4);
    float* agg  = (float*)(ws + off);       off += align256((size_t)N * H * 4);
    bf16_t* w1t = (bf16_t*)(ws + off);      off += align256((size_t)H * IN * 2);
    bf16_t* w2t = (bf16_t*)(ws + off);      off += align256((size_t)H * H * 2);
    bf16_t* wlt = (bf16_t*)(ws + off);      off += align256((size_t)OUT * H * 2);

    const int T = 256;
    const int mtiles  = N / 16;                       // 3125
    const int mblocks = (mtiles + BM_TILES - 1) / BM_TILES;  // 391

    // 1) normalization coefficients
    deg_init_kernel<<<(N + T - 1) / T, T, 0, stream>>>(dinv, N);
    deg_accum_kernel<<<2048, T, 0, stream>>>(dst, dinv, E);
    deg_rsqrt_kernel<<<(N + T - 1) / T, T, 0, stream>>>(dinv, N);

    // 2) pack operands to bf16
    cvt_bf16_kernel<<<4096, T, 0, stream>>>(x, xbf, N * IN);
    pack_wT_kernel<<<512, T, 0, stream>>>(W1, w1t, IN, H);
    pack_wT_kernel<<<1024, T, 0, stream>>>(W2, w2t, H, H);
    pack_wT_kernel<<<128, T, 0, stream>>>(Wl, wlt, H, OUT);

    // 3) layer 1: GEMM (N x IN @ IN x H) -> aggregate -> relu
    wmma_gemm_blocked_kernel<<<mblocks * (H / BN), T, 0, stream>>>(
        xbf, w1t, gout, nullptr, N, H, IN);
    agg_init_kernel<<<8192, T, 0, stream>>>(gout, dinv, agg, N, H);
    edge_scatter_kernel<<<32768, T, 0, stream>>>(gout, src, dst, dinv, agg, E, H);
    finalize_kernel<<<8192, T, 0, stream>>>(agg, b1, hbf, N, H);

    // 4) layer 2: GEMM (N x H @ H x H) -> aggregate -> relu
    wmma_gemm_blocked_kernel<<<mblocks * (H / BN), T, 0, stream>>>(
        hbf, w2t, gout, nullptr, N, H, H);
    agg_init_kernel<<<8192, T, 0, stream>>>(gout, dinv, agg, N, H);
    edge_scatter_kernel<<<32768, T, 0, stream>>>(gout, src, dst, dinv, agg, E, H);
    finalize_kernel<<<8192, T, 0, stream>>>(agg, b2, hbf, N, H);

    // 5) output projection: (N x H @ H x OUT) + bl -> d_out (f32)
    wmma_gemm_blocked_kernel<<<mblocks * (OUT / BN), T, 0, stream>>>(
        hbf, wlt, out, bl, N, OUT, H);
}
